// TreeMessageEncoder_75273596829885
// MI455X (gfx1250) — matrix operand order repaired and checked by reading, not compiled, not run on hardware
//
#include <hip/hip_runtime.h>
#include <hip/hip_fp16.h>
#include <math.h>

typedef __attribute__((ext_vector_type(16))) _Float16 v16h;
typedef __attribute__((ext_vector_type(8)))  _Float16 v8h;
typedef __attribute__((ext_vector_type(8)))  float    v8f;

#define BB 64
#define NN 13
#define SS 128
#define HH 768
#define DD 512
#define MROWS (BB * NN)   // 832

__device__ __forceinline__ float gelu_erf(float x) {
    return 0.5f * x * (1.0f + erff(x * 0.7071067811865475f));
}

// ---- wave-level 16x16 output tile ----
// C = A(f16, MxK row-major) @ B, with B supplied TRANSPOSED: Bt is (N x K) row-major f16.
// All fragment loads are contiguous 16-byte vectors per lane.
template <int K, int N>
__device__ __forceinline__ v8f wmma_tile(const _Float16* __restrict__ A,
                                         const _Float16* __restrict__ Bt,
                                         int m0, int n0) {
    const int lane = threadIdx.x & 31;
    const int hi   = lane >> 4;    // half-wave select
    const int l    = lane & 15;
    v8f acc = {};
    const _Float16* arow = A  + (size_t)(m0 + l) * K + hi * 8;   // A frag: K = k0+hi*8+j / +16
    const _Float16* brow = Bt + (size_t)(n0 + l) * K + hi * 8;   // B frag: K = k0+hi*8+j / +16
#pragma unroll 4
    for (int k0 = 0; k0 < K; k0 += 32) {
        union { v16h v; v8h h[2]; } a, b;
        a.h[0] = *(const v8h*)(arow + k0);
        a.h[1] = *(const v8h*)(arow + k0 + 16);
        b.h[0] = *(const v8h*)(brow + k0);
        b.h[1] = *(const v8h*)(brow + k0 + 16);
        acc = __builtin_amdgcn_wmma_f32_16x16x32_f16(false, a.v, false, b.v,
                                                     (short)0, acc, false, false);
    }
    return acc;
}

// ---- f32 (K x N) -> f16 transposed (N x K) for weights ----
__global__ void k_cvt_t(const float* __restrict__ src, _Float16* __restrict__ dst,
                        int K, int N) {
    int i = blockIdx.x * blockDim.x + threadIdx.x;
    if (i < K * N) {
        int k = i / N, n = i % N;
        dst[(size_t)n * K + k] = (_Float16)src[i];
    }
}

// ---- masked mean pool: one block per (b,n); 256 threads cover H=768 (3 each) ----
__global__ void __launch_bounds__(256)
k_pool(const float* __restrict__ leaf_states, const int* __restrict__ leaf_masks,
       _Float16* __restrict__ pooled_h) {
    const int bn = blockIdx.x;               // 0..831
    const int t  = threadIdx.x;
    const float* xs = leaf_states + (size_t)bn * SS * HH;
    const int*   ms = leaf_masks + (size_t)bn * SS;
    float a0 = 0.f, a1 = 0.f, a2 = 0.f, cnt = 0.f;
    for (int s = 0; s < SS; ++s) {
        const int m = ms[s];                 // uniform across the block
        cnt += (m != 0) ? 1.f : 0.f;
        if (m != 0) {
            const float* row = xs + (size_t)s * HH;
            a0 += row[t]; a1 += row[t + 256]; a2 += row[t + 512];
        }
    }
    const float inv = 1.f / fmaxf(cnt, 1.f);
    _Float16* o = pooled_h + (size_t)bn * HH;
    o[t]       = (_Float16)(a0 * inv);
    o[t + 256] = (_Float16)(a1 * inv);
    o[t + 512] = (_Float16)(a2 * inv);
}

// ---- leaf layer 1: H1 = gelu(pooled @ Wl1 + bl1), f16 out. 52x32 tiles = 1664 waves ----
__global__ void __launch_bounds__(256)
k_leaf1(const _Float16* __restrict__ Ah, const _Float16* __restrict__ Wt,
        const float* __restrict__ bias, _Float16* __restrict__ H1h) {
    const int wave = (blockIdx.x * blockDim.x + threadIdx.x) >> 5;
    const int m0 = (wave >> 5) * 16;         // 32 n-tiles
    const int n0 = (wave & 31) * 16;
    v8f acc = wmma_tile<HH, DD>(Ah, Wt, m0, n0);
    const int lane = threadIdx.x & 31, hi = lane >> 4, l = lane & 15;
#pragma unroll
    for (int r = 0; r < 8; ++r) {
        const int mm = m0 + r + hi * 8, nn = n0 + l;
        H1h[(size_t)mm * DD + nn] = (_Float16)gelu_erf(acc[r] + bias[nn]);
    }
}

// ---- leaf layer 2 + depth embed + leaf mask -> node_repr (f32 + f16 mirror) ----
__global__ void __launch_bounds__(256)
k_leaf2(const _Float16* __restrict__ H1h, const _Float16* __restrict__ Wt,
        const float* __restrict__ bias, const int* __restrict__ depth,
        const int* __restrict__ is_leaf, const float* __restrict__ depth_emb,
        float* __restrict__ node_f32, _Float16* __restrict__ node_h) {
    const int wave = (blockIdx.x * blockDim.x + threadIdx.x) >> 5;
    const int m0 = (wave >> 5) * 16;
    const int n0 = (wave & 31) * 16;
    v8f acc = wmma_tile<DD, DD>(H1h, Wt, m0, n0);
    const int lane = threadIdx.x & 31, hi = lane >> 4, l = lane & 15;
#pragma unroll
    for (int r = 0; r < 8; ++r) {
        const int mm = m0 + r + hi * 8, nn = n0 + l;
        const int node = mm % NN;
        float v = acc[r] + bias[nn] + depth_emb[(size_t)depth[node] * DD + nn];
        v *= (is_leaf[mm] != 0) ? 1.f : 0.f;
        node_f32[(size_t)mm * DD + nn] = v;
        node_h[(size_t)mm * DD + nn]   = (_Float16)v;
    }
}

// ---- merge prep: cat children (f16) + any(is_internal) flag. Grid = 64 (per batch) ----
__global__ void __launch_bounds__(256)
k_merge_prep(int node, const int* __restrict__ left, const int* __restrict__ right,
             const int* __restrict__ active, const int* __restrict__ is_leaf,
             const float* __restrict__ node_f32, _Float16* __restrict__ cat_h,
             int* __restrict__ flag) {
    const int lc = left[node];
    if (lc < 0) return;
    const int rc = right[node];
    const int b = blockIdx.x;
    for (int c = threadIdx.x; c < 2 * DD; c += 256) {
        const float v = (c < DD) ? node_f32[((size_t)(b * NN + lc)) * DD + c]
                                 : node_f32[((size_t)(b * NN + rc)) * DD + (c - DD)];
        cat_h[(size_t)b * (2 * DD) + c] = (_Float16)v;
    }
    if (b == 0 && threadIdx.x == 0) {
        int any = 0;
        for (int bb = 0; bb < BB; ++bb)
            any |= ((active[bb * NN + node] != 0) && (is_leaf[bb * NN + node] == 0)) ? 1 : 0;
        *flag = any;
    }
}

// ---- merge GEMM1: gelu(cat @ Wm1 + bm1) -> Hm (f16). 4x32 tiles = 128 waves ----
__global__ void __launch_bounds__(256)
k_merge_g1(int node, const int* __restrict__ left, const _Float16* __restrict__ cat_h,
           const _Float16* __restrict__ Wt, const float* __restrict__ bias,
           _Float16* __restrict__ Hm_h) {
    if (left[node] < 0) return;
    const int wave = (blockIdx.x * blockDim.x + threadIdx.x) >> 5;
    const int m0 = (wave >> 5) * 16;
    const int n0 = (wave & 31) * 16;
    v8f acc = wmma_tile<2 * DD, DD>(cat_h, Wt, m0, n0);
    const int lane = threadIdx.x & 31, hi = lane >> 4, l = lane & 15;
#pragma unroll
    for (int r = 0; r < 8; ++r) {
        const int mm = m0 + r + hi * 8, nn = n0 + l;
        Hm_h[(size_t)mm * DD + nn] = (_Float16)gelu_erf(acc[r] + bias[nn]);
    }
}

// ---- merge GEMM2 + epilogue: masked column overwrite of node_repr[:, node] ----
__global__ void __launch_bounds__(256)
k_merge_g2(int node, const int* __restrict__ left, const int* __restrict__ active,
           const int* __restrict__ is_leaf, const int* __restrict__ depth,
           const _Float16* __restrict__ Hm_h, const _Float16* __restrict__ Wt,
           const float* __restrict__ bias, const float* __restrict__ depth_emb,
           const int* __restrict__ flag,
           float* __restrict__ node_f32, _Float16* __restrict__ node_h) {
    if (left[node] < 0) return;
    if (*flag == 0) return;                  // keep old column, matches jnp.where
    const int wave = (blockIdx.x * blockDim.x + threadIdx.x) >> 5;
    const int m0 = (wave >> 5) * 16;         // M = 64 rows (batch)
    const int n0 = (wave & 31) * 16;
    v8f acc = wmma_tile<DD, DD>(Hm_h, Wt, m0, n0);
    const int lane = threadIdx.x & 31, hi = lane >> 4, l = lane & 15;
    const int dp = depth[node];
#pragma unroll
    for (int r = 0; r < 8; ++r) {
        const int b = m0 + r + hi * 8, nn = n0 + l;
        float v = acc[r] + bias[nn] + depth_emb[(size_t)dp * DD + nn];
        const int isint = (active[b * NN + node] != 0) && (is_leaf[b * NN + node] == 0);
        v *= isint ? 1.f : 0.f;
        node_f32[((size_t)(b * NN + node)) * DD + nn] = v;
        node_h[((size_t)(b * NN + node)) * DD + nn]   = (_Float16)v;
    }
}

// ---- final: topology hash -> shape embedding + root column ----
__global__ void __launch_bounds__(256)
k_final(const int* __restrict__ active, const int* __restrict__ is_leaf,
        const float* __restrict__ node_f32, const float* __restrict__ shape_emb,
        float* __restrict__ out) {
    const int b = blockIdx.x;
    long long hsum = 0, w = 1;
    for (int n = 0; n < NN; ++n) {
        const long long pat = (long long)((active[b * NN + n] != 0) ? 2 : 0)
                            + (long long)((is_leaf[b * NN + n] != 0) ? 1 : 0);
        hsum += pat * w;
        w *= 31;
    }
    if (hsum < 0) hsum = -hsum;
    const int hid = (int)(hsum % 256);
    for (int n = threadIdx.x; n < DD; n += 256)
        out[(size_t)b * DD + n] = node_f32[((size_t)(b * NN)) * DD + n]
                                + shape_emb[(size_t)hid * DD + n];
}

static inline size_t align256(size_t x) { return (x + 255) & ~(size_t)255; }

extern "C" void kernel_launch(void* const* d_in, const int* in_sizes, int n_in,
                              void* d_out, int out_size, void* d_ws, size_t ws_size,
                              hipStream_t stream) {
    const int*   is_leaf     = (const int*)d_in[0];
    const int*   active      = (const int*)d_in[1];
    const int*   depth       = (const int*)d_in[2];
    const int*   left        = (const int*)d_in[3];
    const int*   right       = (const int*)d_in[4];
    const float* leaf_states = (const float*)d_in[5];
    const int*   leaf_masks  = (const int*)d_in[6];
    const float* Wl1 = (const float*)d_in[7];
    const float* bl1 = (const float*)d_in[8];
    const float* Wl2 = (const float*)d_in[9];
    const float* bl2 = (const float*)d_in[10];
    const float* Wm1 = (const float*)d_in[11];
    const float* bm1 = (const float*)d_in[12];
    const float* Wm2 = (const float*)d_in[13];
    const float* bm2 = (const float*)d_in[14];
    const float* depth_emb = (const float*)d_in[15];
    const float* shape_emb = (const float*)d_in[16];
    float* out = (float*)d_out;

    // workspace carve-up (all 256B aligned); weight buffers hold TRANSPOSED f16 (N x K)
    char* p = (char*)d_ws;
    size_t off = 0;
    _Float16* pooled_h = (_Float16*)(p + off); off = align256(off + (size_t)MROWS * HH * 2);
    _Float16* Wl1t     = (_Float16*)(p + off); off = align256(off + (size_t)HH * DD * 2);
    _Float16* Wl2t     = (_Float16*)(p + off); off = align256(off + (size_t)DD * DD * 2);
    _Float16* Wm1t     = (_Float16*)(p + off); off = align256(off + (size_t)2 * DD * DD * 2);
    _Float16* Wm2t     = (_Float16*)(p + off); off = align256(off + (size_t)DD * DD * 2);
    _Float16* H1h      = (_Float16*)(p + off); off = align256(off + (size_t)MROWS * DD * 2);
    float*    node_f32 = (float*)   (p + off); off = align256(off + (size_t)MROWS * DD * 4);
    _Float16* node_h   = (_Float16*)(p + off); off = align256(off + (size_t)MROWS * DD * 2);
    _Float16* cat_h    = (_Float16*)(p + off); off = align256(off + (size_t)BB * 2 * DD * 2);
    _Float16* Hm_h     = (_Float16*)(p + off); off = align256(off + (size_t)BB * DD * 2);
    int*      flag     = (int*)     (p + off); off = align256(off + 256);
    (void)ws_size; (void)n_in; (void)in_sizes; (void)out_size;

    // weight conversions to transposed f16
    k_cvt_t<<<(HH * DD + 255) / 256, 256, 0, stream>>>(Wl1, Wl1t, HH, DD);
    k_cvt_t<<<(DD * DD + 255) / 256, 256, 0, stream>>>(Wl2, Wl2t, DD, DD);
    k_cvt_t<<<(2 * DD * DD + 255) / 256, 256, 0, stream>>>(Wm1, Wm1t, 2 * DD, DD);
    k_cvt_t<<<(DD * DD + 255) / 256, 256, 0, stream>>>(Wm2, Wm2t, DD, DD);

    // pooling (bandwidth-dominant pass): one block per (b, node)
    k_pool<<<MROWS, 256, 0, stream>>>(leaf_states, leaf_masks, pooled_h);

    // leaf MLP: 52x32 tiles -> 1664 waves -> 208 blocks of 8 waves
    k_leaf1<<<208, 256, 0, stream>>>(pooled_h, Wl1t, bl1, H1h);
    k_leaf2<<<208, 256, 0, stream>>>(H1h, Wl2t, bl2, depth, is_leaf, depth_emb,
                                     node_f32, node_h);

    // bottom-up merge, reverse BFS; leaf nodes no-op inside kernels
    for (int i = NN - 1; i >= 0; --i) {
        k_merge_prep<<<BB, 256, 0, stream>>>(i, left, right, active, is_leaf,
                                             node_f32, cat_h, flag);
        k_merge_g1<<<16, 256, 0, stream>>>(i, left, cat_h, Wm1t, bm1, Hm_h);
        k_merge_g2<<<16, 256, 0, stream>>>(i, left, active, is_leaf, depth,
                                           Hm_h, Wm2t, bm2, depth_emb, flag,
                                           node_f32, node_h);
    }

    // hash + shape embedding + root readout
    k_final<<<BB, 256, 0, stream>>>(active, is_leaf, node_f32, shape_emb, out);
}
